// Attention_16844861735369
// MI455X (gfx1250) — compile-verified, matrix operand order checked by reading, not run on hardware
//
#include <hip/hip_runtime.h>
#include <hip/hip_bf16.h>
#include <stdint.h>

typedef __attribute__((ext_vector_type(16))) _Float16 v16h;
typedef __attribute__((ext_vector_type(8)))  _Float16 v8h;
typedef __attribute__((ext_vector_type(4)))  _Float16 v4h;
typedef __attribute__((ext_vector_type(8)))  float    v8f;
typedef __attribute__((ext_vector_type(4)))  int      v4i;

#define B_   4
#define S_   4096
#define WD_  512
#define ED_  64

union V16H { v16h v; v8h p[2]; _Float16 h[16]; };
union V8F  { v8f  v; float f[8]; };
union U32H { uint32_t u; _Float16 h[2]; };

// --- optional gfx1250 async global->LDS path (ASYNCcnt-tracked) ------------
#if defined(__has_builtin)
#if __has_builtin(__builtin_amdgcn_global_load_async_to_lds_b128)
#define HAVE_ASYNC_LDS 1
#endif
#if __has_builtin(__builtin_amdgcn_s_wait_asynccnt)
#define HAVE_WAIT_ASYNC_BUILTIN 1
#endif
#endif

__device__ __forceinline__ void wait_asynccnt0() {
#if defined(HAVE_ASYNC_LDS)
#if defined(HAVE_WAIT_ASYNC_BUILTIN)
    __builtin_amdgcn_s_wait_asynccnt(0);
#else
    asm volatile("s_wait_asynccnt 0" ::: "memory");
#endif
#endif
}

// ---------------------------------------------------------------------------
// Kernel 0: convert W matrices to f16, transposed [col][k], so B-operand
// reads in the projection kernel are contiguous b128 loads.
// ---------------------------------------------------------------------------
__global__ __launch_bounds__(256)
void wcvt_kernel(const float* __restrict__ Wq,
                 const float* __restrict__ Wk,
                 const float* __restrict__ Wv,
                 _Float16* __restrict__ WtQ,
                 _Float16* __restrict__ WtK,
                 _Float16* __restrict__ WtV)
{
    const int mat   = blockIdx.x >> 7;                       // 128 blocks per mat
    const int local = ((blockIdx.x & 127) << 8) + threadIdx.x;  // 0..32767
    const float* W  = (mat == 0) ? Wq  : (mat == 1) ? Wk  : Wv;
    _Float16*   Wt  = (mat == 0) ? WtQ : (mat == 1) ? WtK : WtV;
    const int r = local >> 6;          // 0..511  (k index)
    const int c = local & 63;          // 0..63   (output col)
    Wt[(size_t)c * WD_ + r] = (_Float16)W[(size_t)r * ED_ + c];
}

// ---------------------------------------------------------------------------
// Kernel 1: QKV projection. One block owns 32 x-rows, staged once to LDS as
// f16 (x read from HBM exactly once). Each wave computes the Q, K and V
// 16x16 tiles for its (row-tile, col-tile): shared A operand, 3 WMMAs/step.
// ---------------------------------------------------------------------------
__global__ __launch_bounds__(256)
void qkv_proj_kernel(const float* __restrict__ x,
                     const _Float16* __restrict__ WtQ,
                     const _Float16* __restrict__ WtK,
                     const _Float16* __restrict__ WtV,
                     const float* __restrict__ bqp,
                     const float* __restrict__ bkp,
                     const float* __restrict__ bvp,
                     _Float16* __restrict__ Qh,
                     _Float16* __restrict__ Kh,
                     _Float16* __restrict__ Vh)
{
    __shared__ __align__(16) _Float16 Xs[32 * WD_];   // 32 KB

    const int tid  = threadIdx.x;
    const int lane = tid & 31;
    const int wave = tid >> 5;
    const int hi   = lane >> 4;
    const int n    = lane & 15;
    const int bb   = blockIdx.x;                      // 512 blocks, 32 rows each

    // ---- stage 32x512 f32 -> f16 into LDS (each thread: 16 float4) -------
    {
        const float4* xsrc = (const float4*)(x + (size_t)bb * 32 * WD_);
        v4h* xdst = (v4h*)Xs;
        #pragma unroll
        for (int i = 0; i < 16; ++i) {
            const int idx = tid + i * 256;
            const float4 f = xsrc[idx];
            v4h h;
            h.x = (_Float16)f.x; h.y = (_Float16)f.y;
            h.z = (_Float16)f.z; h.w = (_Float16)f.w;
            xdst[idx] = h;
        }
    }
    __syncthreads();

    const int lrow0 = (wave >> 2) * 16;               // 0 or 16
    const int n0    = (wave & 3) * 16;

    const _Float16* xs = Xs  + (size_t)(lrow0 + n) * WD_ + (hi << 3);
    const _Float16* wq = WtQ + (size_t)(n0 + n)  * WD_ + (hi << 3);
    const _Float16* wk = WtK + (size_t)(n0 + n)  * WD_ + (hi << 3);
    const _Float16* wv = WtV + (size_t)(n0 + n)  * WD_ + (hi << 3);

    V8F cq, ck, cv;
    #pragma unroll
    for (int r = 0; r < 8; ++r) { cq.f[r] = 0.0f; ck.f[r] = 0.0f; cv.f[r] = 0.0f; }

    #pragma unroll
    for (int kk = 0; kk < WD_; kk += 32) {
        V16H a, wbq, wbk, wbv;
        a.p[0]   = *(const v8h*)(xs + kk);  a.p[1]   = *(const v8h*)(xs + kk + 16);
        wbq.p[0] = *(const v8h*)(wq + kk);  wbq.p[1] = *(const v8h*)(wq + kk + 16);
        wbk.p[0] = *(const v8h*)(wk + kk);  wbk.p[1] = *(const v8h*)(wk + kk + 16);
        wbv.p[0] = *(const v8h*)(wv + kk);  wbv.p[1] = *(const v8h*)(wv + kk + 16);
        cq.v = __builtin_amdgcn_wmma_f32_16x16x32_f16(false, a.v, false, wbq.v, (short)0, cq.v, false, false);
        ck.v = __builtin_amdgcn_wmma_f32_16x16x32_f16(false, a.v, false, wbk.v, (short)0, ck.v, false, false);
        cv.v = __builtin_amdgcn_wmma_f32_16x16x32_f16(false, a.v, false, wbv.v, (short)0, cv.v, false, false);
    }

    const float biasq = bqp[n0 + n];
    const float biask = bkp[n0 + n];
    const float biasv = bvp[n0 + n];
    #pragma unroll
    for (int r = 0; r < 8; ++r) {
        const int orow = bb * 32 + lrow0 + r + (hi << 3);
        Qh[(size_t)orow * ED_ + n0 + n] = (_Float16)(cq.f[r] + biasq);
        Kh[(size_t)orow * ED_ + n0 + n] = (_Float16)(ck.f[r] + biask);
        Vh[(size_t)orow * ED_ + n0 + n] = (_Float16)(cv.f[r] + biasv);
    }
}

// ---------------------------------------------------------------------------
// Kernel 2: flash attention. 8 waves/block, wave owns 16 query rows.
// K tile staged to LDS via async-to-LDS, V staged transposed so every
// WMMA operand is built from b128 loads.
// ---------------------------------------------------------------------------
__global__ __launch_bounds__(256)
void flash_attn_kernel(const _Float16* __restrict__ Qh,
                       const _Float16* __restrict__ Kh,
                       const _Float16* __restrict__ Vh,
                       float* __restrict__ out)
{
    __shared__ __align__(16) _Float16 Kt [32 * ED_];      // row-major [key][dim]
    __shared__ __align__(16) _Float16 VtT[ED_ * 32];      // transposed [dim][key]
    __shared__ __align__(16) _Float16 Pt [8 * 16 * 32];   // per-wave 16x32 P patch

    const int tid  = threadIdx.x;
    const int lane = tid & 31;
    const int wave = tid >> 5;
    const int hi   = lane >> 4;
    const int n    = lane & 15;

    const int b  = blockIdx.x >> 5;
    const int qb = blockIdx.x & 31;
    const int g0 = b * S_ + qb * 128 + wave * 16;

    V16H qa0, qa1;
    {
        const _Float16* gq = Qh + (size_t)(g0 + n) * ED_ + (hi << 3);
        qa0.p[0] = *(const v8h*)(gq);
        qa0.p[1] = *(const v8h*)(gq + 16);
        qa1.p[0] = *(const v8h*)(gq + 32);
        qa1.p[1] = *(const v8h*)(gq + 48);
    }

    V8F acc[4];
    float m_i[8], l_i[8];
    #pragma unroll
    for (int r = 0; r < 8; ++r) { m_i[r] = -3.0e38f; l_i[r] = 0.0f; }
    #pragma unroll
    for (int t = 0; t < 4; ++t)
        #pragma unroll
        for (int r = 0; r < 8; ++r) acc[t].f[r] = 0.0f;

    const _Float16* Kbase = Kh + (size_t)b * S_ * ED_;
    const _Float16* Vbase = Vh + (size_t)b * S_ * ED_;
    const float scale = 0.125f;   // 1/sqrt(64)

    for (int kt = 0; kt < S_ / 32; ++kt) {
        const _Float16* ks = Kbase + (size_t)kt * 32 * ED_;   // 4 KB tile
        const _Float16* vs = Vbase + (size_t)kt * 32 * ED_;

        // ---- K tile: 256 threads x 16B, direct-to-LDS if available --------
#if defined(HAVE_ASYNC_LDS)
        __builtin_amdgcn_global_load_async_to_lds_b128(
            (v4i*)((const char*)ks + tid * 16),
            (v4i*)((char*)Kt + tid * 16), 0, 0);
#else
        *(float4*)((char*)Kt + tid * 16) =
            *(const float4*)((const char*)ks + tid * 16);
#endif
        // ---- V tile: stage transposed [dim][key] so B-reads are contiguous
        {
            const uint32_t* vs32 = (const uint32_t*)vs;
            #pragma unroll
            for (int i = 0; i < 4; ++i) {
                const int idx = tid + i * 256;        // 0..1023
                U32H cvt; cvt.u = vs32[idx];
                const int k = idx >> 5;               // key 0..31
                const int d = (idx & 31) << 1;        // dim 0..62
                VtT[d * 32 + k]       = cvt.h[0];
                VtT[(d + 1) * 32 + k] = cvt.h[1];
            }
        }
        if (kt + 1 < S_ / 32) {                       // L2 prefetch next tiles
            __builtin_prefetch((const char*)ks + 4096, 0, 0);
            __builtin_prefetch((const char*)vs + 4096, 0, 0);
        }
        wait_asynccnt0();
        __syncthreads();

        // ---- scores: two 16-key subtiles, inner dim 64 = 2 WMMAs each -----
        V8F s0, s1;
        {
            const _Float16* kb = Kt + (size_t)n * ED_ + (hi << 3);
            V16H kb0, kb1;
            kb0.p[0] = *(const v8h*)(kb);        kb0.p[1] = *(const v8h*)(kb + 16);
            kb1.p[0] = *(const v8h*)(kb + 32);   kb1.p[1] = *(const v8h*)(kb + 48);
            v8f z = {};
            z = __builtin_amdgcn_wmma_f32_16x16x32_f16(false, qa0.v, false, kb0.v, (short)0, z, false, false);
            z = __builtin_amdgcn_wmma_f32_16x16x32_f16(false, qa1.v, false, kb1.v, (short)0, z, false, false);
            s0.v = z;
        }
        {
            const _Float16* kb = Kt + (size_t)(16 + n) * ED_ + (hi << 3);
            V16H kb0, kb1;
            kb0.p[0] = *(const v8h*)(kb);        kb0.p[1] = *(const v8h*)(kb + 16);
            kb1.p[0] = *(const v8h*)(kb + 32);   kb1.p[1] = *(const v8h*)(kb + 48);
            v8f z = {};
            z = __builtin_amdgcn_wmma_f32_16x16x32_f16(false, qa0.v, false, kb0.v, (short)0, z, false, false);
            z = __builtin_amdgcn_wmma_f32_16x16x32_f16(false, qa1.v, false, kb1.v, (short)0, z, false, false);
            s1.v = z;
        }

        // ---- online softmax (lanes 0-15 own rows 0-7, 16-31 rows 8-15) ----
        float alpha[8];
        #pragma unroll
        for (int r = 0; r < 8; ++r) {
            const float a0 = s0.f[r] * scale;
            const float a1 = s1.f[r] * scale;
            float mx = fmaxf(a0, a1);
            #pragma unroll
            for (int msk = 8; msk >= 1; msk >>= 1)
                mx = fmaxf(mx, __shfl_xor(mx, msk, 32));
            const float mnew = fmaxf(m_i[r], mx);
            const float p0 = __expf(a0 - mnew);
            const float p1 = __expf(a1 - mnew);
            float rs = p0 + p1;
            #pragma unroll
            for (int msk = 8; msk >= 1; msk >>= 1)
                rs += __shfl_xor(rs, msk, 32);
            alpha[r] = __expf(m_i[r] - mnew);
            l_i[r]   = l_i[r] * alpha[r] + rs;
            m_i[r]   = mnew;
            const int m = r + (hi << 3);
            Pt[wave * 512 + m * 32 + n]      = (_Float16)p0;
            Pt[wave * 512 + m * 32 + 16 + n] = (_Float16)p1;
        }

        #pragma unroll
        for (int t = 0; t < 4; ++t)
            #pragma unroll
            for (int r = 0; r < 8; ++r) acc[t].f[r] *= alpha[r];

        // wave-private LDS patch: hardware keeps same-wave DS ops in order,
        // just stop the compiler reordering stores vs. reloads.
        asm volatile("" ::: "memory");

        // ---- reload P in A layout: two b128 loads ----
        V16H pa;
        {
            const _Float16* pp = Pt + wave * 512 + n * 32 + (hi << 3);
            pa.p[0] = *(const v8h*)(pp);
            pa.p[1] = *(const v8h*)(pp + 16);
        }

        // ---- P @ V: V transposed in LDS -> contiguous b128 reads ----------
        #pragma unroll
        for (int t = 0; t < 4; ++t) {
            const _Float16* vv = VtT + (size_t)(t * 16 + n) * 32 + (hi << 3);
            V16H vb;
            vb.p[0] = *(const v8h*)(vv);
            vb.p[1] = *(const v8h*)(vv + 16);
            acc[t].v = __builtin_amdgcn_wmma_f32_16x16x32_f16(
                           false, pa.v, false, vb.v, (short)0, acc[t].v, false, false);
        }

        __syncthreads();   // protect Kt/VtT before next staging pass
    }

    #pragma unroll
    for (int r = 0; r < 8; ++r) {
        const float inv  = 1.0f / l_i[r];
        const int   orow = g0 + r + (hi << 3);
        #pragma unroll
        for (int t = 0; t < 4; ++t)
            out[(size_t)orow * ED_ + t * 16 + n] = acc[t].f[r] * inv;
    }
}

// ---------------------------------------------------------------------------
extern "C" void kernel_launch(void* const* d_in, const int* in_sizes, int n_in,
                              void* d_out, int out_size, void* d_ws, size_t ws_size,
                              hipStream_t stream)
{
    const float* x  = (const float*)d_in[0];
    const float* Wq = (const float*)d_in[1];
    const float* bq = (const float*)d_in[2];
    const float* Wk = (const float*)d_in[3];
    const float* bk = (const float*)d_in[4];
    const float* Wv = (const float*)d_in[5];
    const float* bv = (const float*)d_in[6];
    float* out = (float*)d_out;

    // workspace: Q/K/V f16 (3 x 2 MB) + transposed f16 weights (3 x 64 KB)
    _Float16* Qh  = (_Float16*)d_ws;
    _Float16* Kh  = Qh + (size_t)B_ * S_ * ED_;
    _Float16* Vh  = Kh + (size_t)B_ * S_ * ED_;
    _Float16* WtQ = Vh + (size_t)B_ * S_ * ED_;
    _Float16* WtK = WtQ + (size_t)WD_ * ED_;
    _Float16* WtV = WtK + (size_t)WD_ * ED_;

    wcvt_kernel<<<384, 256, 0, stream>>>(Wq, Wk, Wv, WtQ, WtK, WtV);
    qkv_proj_kernel<<<512, 256, 0, stream>>>(x, WtQ, WtK, WtV, bq, bk, bv, Qh, Kh, Vh);
    flash_attn_kernel<<<128, 256, 0, stream>>>(Qh, Kh, Vh, out);
}
